// Trinity_Net_10840497455435
// MI455X (gfx1250) — matrix operand order repaired and checked by reading, not compiled, not run on hardware
//
#include <hip/hip_runtime.h>

// ---------------------------------------------------------------------------
// Swin block for MI455X (gfx1250, wave32, WMMA 16x16x32 bf16, f32 accum)
// B=4 H=W=256 C=96 WS=8 SS=4 NH=3 N=64 NW=1024 HD=32 MLP_H=384
// Weights pre-swizzled into per-lane WMMA B-fragment order: one 32B load/lane.
// ---------------------------------------------------------------------------

typedef __bf16 bf16;
typedef __bf16 v16bf __attribute__((ext_vector_type(16)));
typedef __bf16 v8bf  __attribute__((ext_vector_type(8)));
typedef float  v8f   __attribute__((ext_vector_type(8)));

#define WMMA_BF16(a, b, c) \
  __builtin_amdgcn_wmma_f32_16x16x32_bf16(false, (a), false, (b), (short)0, (c), false, false)

// A fragment: 16x32 (MxK) bf16 from a row-major [M][lda] buffer.
// A lane needs K = half*16+0..15 (contiguous order, matching B frags):
// one contiguous 32-byte run -> 2x b128 loads.
__device__ __forceinline__ v16bf load_a_frag(const bf16* base, int lda, int lane) {
  const int m = lane & 15, half = lane >> 4;
  const bf16* p = base + m * lda + (half << 4);
  const v8bf lo = *(const v8bf*)p;
  const v8bf hi = *(const v8bf*)(p + 8);
  return __builtin_shufflevector(lo, hi, 0, 1, 2, 3, 4, 5, 6, 7,
                                 8, 9, 10, 11, 12, 13, 14, 15);
}

// B fragment from swizzled storage: [tile][lane][16 bf16 contiguous].
__device__ __forceinline__ v16bf load_b_swz(const bf16* frag, int lane) {
  return *(const v16bf*)(frag + lane * 16);
}

// ---------------------------------------------------------------------------
// Prep: swizzle weights into fragment order and densify rel-pos bias.
// For B tile (kt,nt), lane l, element e:  n = nt*16+(l&15), k = kt*32+e+((l>>4)<<4).
// Source weights are [N][Korig] row-major.
// ---------------------------------------------------------------------------
__device__ __forceinline__ void swz_store(bf16* __restrict__ dst, const float* __restrict__ w,
                                          int Korig, int Kt, int j) {
  const int e = j & 15;
  const int lane = (j >> 4) & 31;
  const int tile = j >> 9;
  const int kt = tile % Kt;
  const int nt = tile / Kt;
  const int n = nt * 16 + (lane & 15);
  const int k = kt * 32 + e + ((lane >> 4) << 4);
  dst[j] = (bf16)w[n * Korig + k];
}

__global__ void swin_prep_kernel(const float* __restrict__ qkv_w, const float* __restrict__ proj_w,
                                 const float* __restrict__ fc1_w, const float* __restrict__ fc2_w,
                                 const float* __restrict__ rpb, const int* __restrict__ rel,
                                 bf16* __restrict__ wqkv, bf16* __restrict__ wproj,
                                 bf16* __restrict__ wfc1, bf16* __restrict__ wfc2,
                                 float* __restrict__ bias3) {
  int i = blockIdx.x * 256 + threadIdx.x;
  if (i < 27648) { swz_store(wqkv, qkv_w, 96, 3, i);  return; }   // K=96  Kt=3  Nt=18
  i -= 27648;
  if (i < 9216)  { swz_store(wproj, proj_w, 96, 3, i); return; }  // K=96  Kt=3  Nt=6
  i -= 9216;
  if (i < 36864) { swz_store(wfc1, fc1_w, 96, 3, i);  return; }   // K=96  Kt=3  Nt=24
  i -= 36864;
  if (i < 36864) { swz_store(wfc2, fc2_w, 384, 12, i); return; }  // K=384 Kt=12 Nt=6
  i -= 36864;
  if (i < 12288) { int h = i / 4096, rm = i % 4096; bias3[i] = rpb[rel[rm] * 3 + h]; }
}

// ---------------------------------------------------------------------------
// Kernel A: one workgroup (4 wave32) per shifted window.  LDS = 56KB.
// ---------------------------------------------------------------------------
__global__ void __launch_bounds__(128) swin_attn_kernel(
    const float* __restrict__ x, const float* __restrict__ g1, const float* __restrict__ b1,
    const float* __restrict__ qkv_b, const float* __restrict__ proj_b,
    const float* __restrict__ attn_mask,
    const bf16* __restrict__ wqkv, const bf16* __restrict__ wproj,
    const float* __restrict__ bias3, float* __restrict__ out) {
  __shared__ __align__(32) char smem[57344];
  bf16* sX  = (bf16*)smem;              // 64x96 row-major (A operand)
  bf16* sQ  = (bf16*)(smem + 12288);    // [3][64][32] row-major, pre-scaled
  bf16* sK  = (bf16*)(smem + 24576);    // [3][4 key-tiles][32 lanes][16]  swizzled K^T frags
  bf16* sV  = (bf16*)(smem + 36864);    // [3][2 kt][2 nt][32 lanes][16]   swizzled V frags
  bf16* sSP = (bf16*)(smem + 49152);    // [64][64] scores/probs (per head)
  bf16* sO  = sX;                       // 64x96 attention output (sX dead by then)

  const int tid  = threadIdx.x;
  const int wave = tid >> 5, lane = tid & 31;
  const int lo16 = lane & 15, half = lane >> 4;
  const int bb   = blockIdx.x >> 10;
  const int widx = blockIdx.x & 1023;
  const int wi   = widx >> 5, wj = widx & 31;
  const bool masked = (wi == 31) || (wj == 31);   // only boundary windows have a mask

  // ---- LN1 + cyclic-shift window gather (token t owns its 96-ch row) ----
  if (tid < 64) {
    const int r  = tid >> 3, c = tid & 7;
    const int hh = (wi * 8 + r + 4) & 255;
    const int ww = (wj * 8 + c + 4) & 255;
    const float* xr = x + ((size_t)bb * 65536 + (size_t)hh * 256 + ww) * 96;
    float s = 0.f, ss = 0.f;
    for (int ch = 0; ch < 96; ++ch) {
      const float v = xr[ch];
      s += v; ss += v * v;
      sX[tid * 96 + ch] = (bf16)v;
    }
    const float mu   = s * (1.f / 96.f);
    const float var  = ss * (1.f / 96.f) - mu * mu;
    const float rstd = rsqrtf(var + 1e-5f);
    for (int ch = 0; ch < 96; ++ch) {
      const float v = (float)sX[tid * 96 + ch];
      sX[tid * 96 + ch] = (bf16)(((v - mu) * rstd) * g1[ch] + b1[ch]);
    }
  }
  __syncthreads();

  // ---- QKV GEMM: (64x96) @ (96x288); scatter Q(scaled)/K^T/V into frag layouts ----
  for (int t = wave; t < 72; t += 4) {
    const int mt = t & 3, nt = t >> 2;
    v8f acc = {};
#pragma unroll
    for (int kt = 0; kt < 3; ++kt) {
      const v16bf a = load_a_frag(sX + mt * 16 * 96 + kt * 32, 96, lane);
      const v16bf b = load_b_swz(wqkv + (nt * 3 + kt) * 512, lane);
      acc = WMMA_BF16(a, b, acc);
    }
    const float bias = qkv_b[nt * 16 + lo16];
    const int which = nt / 6;          // tile-uniform: 0=Q,1=K,2=V
    const int sub   = nt - which * 6;  // 0..5
    const int h     = sub >> 1;        // tile-uniform head
    const int dhi   = (sub & 1) << 4;  // tile-uniform d bit-4
    if (which == 0) {                  // Q row-major, pre-scaled
#pragma unroll
      for (int e = 0; e < 8; ++e) {
        const int m = mt * 16 + e + (half << 3);
        sQ[(h * 64 + m) * 32 + dhi + lo16] =
            (bf16)((acc[e] + bias) * 0.17677669529663687f);
      }
    } else if (which == 1) {           // K^T swizzled frag: lane'=(m&15)+dhi, elem=d&15
#pragma unroll
      for (int e = 0; e < 8; ++e) {
        const int mlo = e + (half << 3);
        sK[((h * 4 + mt) * 32 + mlo + dhi) * 16 + lo16] = (bf16)(acc[e] + bias);
      }
    } else {                           // V swizzled frag: contiguous 16B store per lane
      const int ktp = mt >> 1, halfp = mt & 1, ntp = sub & 1;
      bf16* dst = sV + (((h * 2 + ktp) * 2 + ntp) * 32 + lo16 + (halfp << 4)) * 16;
      v8bf pv;
#pragma unroll
      for (int e = 0; e < 8; ++e) pv[e] = (bf16)(acc[e] + bias);
      *(v8bf*)(dst + (half << 3)) = pv;
    }
  }
  __syncthreads();

  // ---- per-head attention ----
  for (int h = 0; h < 3; ++h) {
    // S = Q K^T + rel-pos bias (+ shift mask on boundary windows only)
    for (int t = wave; t < 16; t += 4) {
      const int mt = t & 3, nt = t >> 2;
      const v16bf a = load_a_frag(sQ + h * 2048 + mt * 512, 32, lane);
      const v16bf b = load_b_swz(sK + (h * 4 + nt) * 512, lane);
      v8f acc = {};
      acc = WMMA_BF16(a, b, acc);
      const int mcol = nt * 16 + lo16;
#pragma unroll
      for (int e = 0; e < 8; ++e) {
        const int nrow = mt * 16 + e + (half << 3);
        float s = acc[e] + bias3[h * 4096 + nrow * 64 + mcol];
        if (masked) s += attn_mask[(size_t)widx * 4096 + nrow * 64 + mcol];
        sSP[nrow * 64 + mcol] = (bf16)s;
      }
    }
    __syncthreads();
    // softmax (one row per thread, in place)
    if (tid < 64) {
      bf16* row = sSP + tid * 64;
      float mx = -1e30f;
      for (int m = 0; m < 64; ++m) mx = fmaxf(mx, (float)row[m]);
      float sum = 0.f;
      for (int m = 0; m < 64; ++m) sum += __expf((float)row[m] - mx);
      const float rs = 1.f / sum;
      for (int m = 0; m < 64; ++m) row[m] = (bf16)(__expf((float)row[m] - mx) * rs);
    }
    __syncthreads();
    // O_h = P @ V
    for (int t = wave; t < 8; t += 4) {
      const int mt = t & 3, nt = t >> 2;
      v8f acc = {};
#pragma unroll
      for (int kt = 0; kt < 2; ++kt) {
        const v16bf a = load_a_frag(sSP + mt * 16 * 64 + kt * 32, 64, lane);
        const v16bf b = load_b_swz(sV + ((h * 2 + kt) * 2 + nt) * 512, lane);
        acc = WMMA_BF16(a, b, acc);
      }
      const int nd = nt * 16 + lo16;
#pragma unroll
      for (int e = 0; e < 8; ++e)
        sO[(mt * 16 + e + (half << 3)) * 96 + h * 32 + nd] = (bf16)acc[e];
    }
    __syncthreads();
  }

  // ---- proj GEMM + reverse shift scatter + residual ----
  for (int t = wave; t < 24; t += 4) {
    const int mt = t & 3, nt = t >> 2;
    v8f acc = {};
#pragma unroll
    for (int kt = 0; kt < 3; ++kt) {
      const v16bf a = load_a_frag(sO + mt * 16 * 96 + kt * 32, 96, lane);
      const v16bf b = load_b_swz(wproj + (nt * 3 + kt) * 512, lane);
      acc = WMMA_BF16(a, b, acc);
    }
    const int   ncol = nt * 16 + lo16;
    const float pb   = proj_b[ncol];
#pragma unroll
    for (int e = 0; e < 8; ++e) {
      const int m  = mt * 16 + e + (half << 3);
      const int r  = m >> 3, c = m & 7;
      const int hh = (wi * 8 + r + 4) & 255;
      const int ww = (wj * 8 + c + 4) & 255;
      const size_t idx = ((size_t)bb * 65536 + (size_t)hh * 256 + ww) * 96 + ncol;
      out[idx] = x[idx] + acc[e] + pb;
    }
  }
}

// ---------------------------------------------------------------------------
// Kernel B: 64 tokens per workgroup; LN2 -> fc1+GELU -> fc2 -> residual,
// in place on d_out (block exclusively owns its 64 rows).  LDS = 60KB.
// ---------------------------------------------------------------------------
__global__ void __launch_bounds__(128) swin_mlp_kernel(
    const float* __restrict__ g2, const float* __restrict__ b2,
    const float* __restrict__ fc1_b, const float* __restrict__ fc2_b,
    const bf16* __restrict__ wfc1, const bf16* __restrict__ wfc2,
    float* __restrict__ out) {
  __shared__ __align__(32) char smem[61440];
  bf16* sX = (bf16*)smem;             // 64x96
  bf16* sH = (bf16*)(smem + 12288);   // 64x384

  const int    tid  = threadIdx.x;
  const int    wave = tid >> 5, lane = tid & 31;
  const int    lo16 = lane & 15, half = lane >> 4;
  const size_t tok0 = (size_t)blockIdx.x * 64;

  if (tid < 64) {
    const float* xr = out + (tok0 + tid) * 96;
    float s = 0.f, ss = 0.f;
    for (int ch = 0; ch < 96; ++ch) {
      const float v = xr[ch];
      s += v; ss += v * v;
      sX[tid * 96 + ch] = (bf16)v;
    }
    const float mu   = s * (1.f / 96.f);
    const float var  = ss * (1.f / 96.f) - mu * mu;
    const float rstd = rsqrtf(var + 1e-5f);
    for (int ch = 0; ch < 96; ++ch) {
      const float v = (float)sX[tid * 96 + ch];
      sX[tid * 96 + ch] = (bf16)(((v - mu) * rstd) * g2[ch] + b2[ch]);
    }
  }
  __syncthreads();

  // fc1 + exact GELU: (64x96) @ (96x384)
  for (int t = wave; t < 96; t += 4) {
    const int mt = t & 3, nt = t >> 2;
    v8f acc = {};
#pragma unroll
    for (int kt = 0; kt < 3; ++kt) {
      const v16bf a = load_a_frag(sX + mt * 16 * 96 + kt * 32, 96, lane);
      const v16bf b = load_b_swz(wfc1 + (nt * 3 + kt) * 512, lane);
      acc = WMMA_BF16(a, b, acc);
    }
    const int   ncol = nt * 16 + lo16;
    const float bias = fc1_b[ncol];
#pragma unroll
    for (int e = 0; e < 8; ++e) {
      const float v = acc[e] + bias;
      const float g = 0.5f * v * (1.f + erff(v * 0.70710678118654752f));
      sH[(mt * 16 + e + (half << 3)) * 384 + ncol] = (bf16)g;
    }
  }
  __syncthreads();

  // fc2 + residual: (64x384) @ (384x96)
  for (int t = wave; t < 24; t += 4) {
    const int mt = t & 3, nt = t >> 2;
    v8f acc = {};
#pragma unroll
    for (int kt = 0; kt < 12; ++kt) {
      const v16bf a = load_a_frag(sH + mt * 16 * 384 + kt * 32, 384, lane);
      const v16bf b = load_b_swz(wfc2 + (nt * 12 + kt) * 512, lane);
      acc = WMMA_BF16(a, b, acc);
    }
    const int   ncol = nt * 16 + lo16;
    const float bias = fc2_b[ncol];
#pragma unroll
    for (int e = 0; e < 8; ++e) {
      const size_t idx = (tok0 + (size_t)(mt * 16 + e + (half << 3))) * 96 + ncol;
      out[idx] = out[idx] + acc[e] + bias;
    }
  }
}

// ---------------------------------------------------------------------------
extern "C" void kernel_launch(void* const* d_in, const int* in_sizes, int n_in,
                              void* d_out, int out_size, void* d_ws, size_t ws_size,
                              hipStream_t stream) {
  (void)in_sizes; (void)n_in; (void)out_size; (void)ws_size;
  const float* x         = (const float*)d_in[0];
  const float* g1        = (const float*)d_in[1];
  const float* b1        = (const float*)d_in[2];
  const float* qkv_w     = (const float*)d_in[3];
  const float* qkv_b     = (const float*)d_in[4];
  const float* rpb       = (const float*)d_in[5];
  const float* proj_w    = (const float*)d_in[6];
  const float* proj_b    = (const float*)d_in[7];
  const float* g2        = (const float*)d_in[8];
  const float* b2        = (const float*)d_in[9];
  const float* fc1_w     = (const float*)d_in[10];
  const float* fc1_b     = (const float*)d_in[11];
  const float* fc2_w     = (const float*)d_in[12];
  const float* fc2_b     = (const float*)d_in[13];
  const float* attn_mask = (const float*)d_in[14];
  const int*   rel       = (const int*)d_in[15];
  float* out = (float*)d_out;

  char* ws = (char*)d_ws;                 // 270336 bytes used
  bf16*  wqkv  = (bf16*)(ws + 0);         // swizzled [18*3 tiles][32][16]
  bf16*  wproj = (bf16*)(ws + 55296);     // swizzled [6*3  tiles][32][16]
  bf16*  wfc1  = (bf16*)(ws + 73728);     // swizzled [24*3 tiles][32][16]
  bf16*  wfc2  = (bf16*)(ws + 147456);    // swizzled [6*12 tiles][32][16]
  float* bias3 = (float*)(ws + 221184);   // [3][64][64] f32

  hipLaunchKernelGGL(swin_prep_kernel, dim3(480), dim3(256), 0, stream,
                     qkv_w, proj_w, fc1_w, fc2_w, rpb, rel, wqkv, wproj, wfc1, wfc2, bias3);
  hipLaunchKernelGGL(swin_attn_kernel, dim3(4096), dim3(128), 0, stream,
                     x, g1, b1, qkv_b, proj_b, attn_mask, wqkv, wproj, bias3, out);
  hipLaunchKernelGGL(swin_mlp_kernel, dim3(4096), dim3(128), 0, stream,
                     g2, b2, fc1_b, fc2_b, wfc1, wfc2, out);
}
